// Gemma4Model_34952443855136
// MI455X (gfx1250) — matrix operand order repaired and checked by reading, not compiled, not run on hardware
//
#include <hip/hip_runtime.h>
#include <hip/hip_bf16.h>

// ---------------- problem constants (match reference) ----------------
constexpr int T_   = 3072;
constexpr int HID_ = 2048;
constexpr int H_   = 16;
constexpr int KH_  = 8;
constexpr int D_   = 128;
constexpr int WIN_ = 1024;
constexpr int SEQ_ = 1536;
constexpr float EPS_ = 1e-6f;

// ---------------- WMMA fragment types (wave32, gfx1250) ----------------
typedef __attribute__((ext_vector_type(16))) __bf16 v16bf;
typedef __attribute__((ext_vector_type(8)))  __bf16 v8bf;
typedef __attribute__((ext_vector_type(8)))  float  v8f;

// round-to-nearest-even f32 -> bf16 (bit pattern as ushort)
static __device__ __forceinline__ unsigned short f2bf(float f) {
  unsigned int u = __float_as_uint(f);
  u += 0x7FFFu + ((u >> 16) & 1u);
  return (unsigned short)(u >> 16);
}

// concatenate two contiguous 16B bf16 loads into a 16-element fragment
static __device__ __forceinline__ v16bf load_frag(const unsigned short* p0,
                                                  const unsigned short* p1) {
  v8bf lo = *(const v8bf*)p0;
  v8bf hi = *(const v8bf*)p1;
  return __builtin_shufflevector(lo, hi, 0,1,2,3,4,5,6,7,8,9,10,11,12,13,14,15);
}

static __device__ __forceinline__ v8f wmma_bf16(v16bf a, v16bf b, v8f c) {
  return __builtin_amdgcn_wmma_f32_16x16x32_bf16(false, a, false, b,
                                                 (short)0, c, false, false);
}

// ---------------- elementwise cast f32 -> bf16 ----------------
__global__ void cast_f32_bf16_kernel(const float* __restrict__ in,
                                     unsigned short* __restrict__ out, long n) {
  long i = (long)blockIdx.x * blockDim.x + threadIdx.x;
  if (i < n) out[i] = f2bf(in[i]);
}

// ---------------- weight transpose+cast: W[K][N] -> WT[N][K] bf16 ----------------
__global__ void transpose_cast_kernel(const float* __restrict__ W,
                                      unsigned short* __restrict__ WT,
                                      int K, int N) {
  long idx = (long)blockIdx.x * blockDim.x + threadIdx.x;
  if (idx >= (long)K * N) return;
  int k = (int)(idx / N);
  int n = (int)(idx % N);
  WT[(long)n * K + k] = f2bf(W[idx]);
}

// ---------------- GEMM: C[M][N] = A[M][K] @ B[K][N], BT given as [N][K] bf16 ----
// 256 threads = 8 waves; block tile 128x128; wave tile 32x64; K-step 32.
__global__ __launch_bounds__(256) void gemm_bf16_kernel(
    const unsigned short* __restrict__ A,
    const unsigned short* __restrict__ BT,
    float* __restrict__ C, int M, int N, int K) {
  const int lane = threadIdx.x & 31;
  const int half = lane >> 4;
  const int ln   = lane & 15;
  const int wid  = threadIdx.x >> 5;
  const int wm   = wid & 3;   // 4 waves along M
  const int wn   = wid >> 2;  // 2 waves along N
  const long rowbase = (long)blockIdx.y * 128 + wm * 32;
  const long colbase = (long)blockIdx.x * 128 + wn * 64;

  v8f acc[2][4];
#pragma unroll
  for (int mi = 0; mi < 2; ++mi)
#pragma unroll
    for (int ni = 0; ni < 4; ++ni)
#pragma unroll
      for (int r = 0; r < 8; ++r) acc[mi][ni][r] = 0.f;

  for (int kk = 0; kk < K; kk += 32) {
    v16bf a[2], b[4];
#pragma unroll
    for (int mi = 0; mi < 2; ++mi) {
      const unsigned short* p = A + (rowbase + mi * 16 + ln) * K + kk + 8 * half;
      a[mi] = load_frag(p, p + 16);
      __builtin_prefetch(p + 32, 0, 1);   // global_prefetch_b8 for next K-step
    }
#pragma unroll
    for (int ni = 0; ni < 4; ++ni) {
      const unsigned short* p = BT + (colbase + ni * 16 + ln) * K + kk + 8 * half;
      b[ni] = load_frag(p, p + 16);
      __builtin_prefetch(p + 32, 0, 1);
    }
#pragma unroll
    for (int mi = 0; mi < 2; ++mi)
#pragma unroll
      for (int ni = 0; ni < 4; ++ni)
        acc[mi][ni] = wmma_bf16(a[mi], b[ni], acc[mi][ni]);
  }

#pragma unroll
  for (int mi = 0; mi < 2; ++mi)
#pragma unroll
    for (int ni = 0; ni < 4; ++ni)
#pragma unroll
      for (int r = 0; r < 8; ++r) {
        long row = rowbase + mi * 16 + r + 8 * half;
        long col = colbase + ni * 16 + ln;
        C[row * N + col] = acc[mi][ni][r];
      }
}

// ---------------- RMSNorm (+optional gamma) (+optional full-D RoPE) ------------
// X: [T][NH][128] f32 (projection result). One wave per (t,h) row.
// transposed==0: out[(t*NH+h)*128 + d]        (bf16)
// transposed==1: out[(h*128 + d)*T + t]       (bf16)   -> v^T layout for WMMA B
__global__ __launch_bounds__(256) void rms_rope_kernel(
    const float* __restrict__ X, const float* __restrict__ gamma,
    const float* __restrict__ cosb, const float* __restrict__ sinb,
    unsigned short* __restrict__ out, int NH, int do_rope, int transposed) {
  const int wid  = threadIdx.x >> 5;
  const int lane = threadIdx.x & 31;
  const long row = (long)blockIdx.x * 8 + wid;      // over T*NH rows
  const int t = (int)(row / NH);
  const int h = (int)(row % NH);
  const float* xp = X + row * D_;

  float x[4];
#pragma unroll
  for (int j = 0; j < 4; ++j) x[j] = xp[lane + 32 * j];

  float ss = x[0]*x[0] + x[1]*x[1] + x[2]*x[2] + x[3]*x[3];
#pragma unroll
  for (int off = 16; off >= 1; off >>= 1) ss += __shfl_xor(ss, off, 32);
  const float rinv = rsqrtf(ss * (1.0f / D_) + EPS_);

  float y[4];
#pragma unroll
  for (int j = 0; j < 4; ++j) {
    int d = lane + 32 * j;
    float g = gamma ? gamma[d] : 1.0f;
    y[j] = x[j] * rinv * g;
  }

  if (do_rope) {
    float c[4], s[4];
#pragma unroll
    for (int j = 0; j < 4; ++j) {
      int d = lane + 32 * j;
      c[j] = cosb[(long)t * D_ + d];
      s[j] = sinb[(long)t * D_ + d];
    }
    // rotate_half: d<64 partner d+64 (same lane, j+2); d>=64 partner d-64 (j-2)
    float o0 = y[0] * c[0] - y[2] * s[0];
    float o1 = y[1] * c[1] - y[3] * s[1];
    float o2 = y[2] * c[2] + y[0] * s[2];
    float o3 = y[3] * c[3] + y[1] * s[3];
    y[0] = o0; y[1] = o1; y[2] = o2; y[3] = o3;
  }

#pragma unroll
  for (int j = 0; j < 4; ++j) {
    int d = lane + 32 * j;
    if (!transposed) out[row * D_ + d] = f2bf(y[j]);
    else             out[((long)h * D_ + d) * T_ + t] = f2bf(y[j]);
  }
}

// ---------------- flash attention, windowed-causal + segments ------------------
// q: [T][H][128] bf16, k: [T][KH][128] bf16, vT: [KH][128][T] bf16
// One wave per (16-query tile, head). Online softmax over 32-key chunks.
__global__ __launch_bounds__(32) void attn_kernel(
    const unsigned short* __restrict__ qb,
    const unsigned short* __restrict__ kb,
    const unsigned short* __restrict__ vT,
    unsigned short* __restrict__ ob) {
  __shared__ unsigned short pls[16 * 32];   // P tile, row-major [16 q][32 keys]

  const int lane = threadIdx.x & 31;
  const int half = lane >> 4;
  const int ln   = lane & 15;
  const int q0 = blockIdx.x * 16;
  const int h  = blockIdx.y;
  const int kh = h >> 1;                          // GQA: rep = H/KH = 2
  const int seg_start = (q0 >= SEQ_) ? SEQ_ : 0;

  // Q A-fragments over the 4 K-slices of D
  v16bf qf[4];
#pragma unroll
  for (int s = 0; s < 4; ++s) {
    const unsigned short* p = qb + ((long)(q0 + ln) * H_ + h) * D_ + s * 32 + 8 * half;
    qf[s] = load_frag(p, p + 16);
  }

  v8f acc[8];
#pragma unroll
  for (int c = 0; c < 8; ++c)
#pragma unroll
    for (int r = 0; r < 8; ++r) acc[c][r] = 0.f;

  float mrow[8], lrow[8];
#pragma unroll
  for (int r = 0; r < 8; ++r) { mrow[r] = -1e30f; lrow[r] = 0.f; }

  int start = q0 - (WIN_ - 1);
  if (start < seg_start) start = seg_start;
  start &= ~31;

  for (int kc = start; kc < q0 + 16; kc += 32) {
    // S = Q K^T for two 16-key subtiles (f32, C-layout)
    v8f s0 = {0,0,0,0,0,0,0,0};
    v8f s1 = {0,0,0,0,0,0,0,0};
#pragma unroll
    for (int ks = 0; ks < 4; ++ks) {
      const unsigned short* p0 = kb + ((long)(kc + ln) * KH_ + kh) * D_ + ks * 32 + 8 * half;
      const unsigned short* p1 = kb + ((long)(kc + 16 + ln) * KH_ + kh) * D_ + ks * 32 + 8 * half;
      s0 = wmma_bf16(qf[ks], load_frag(p0, p0 + 16), s0);
      s1 = wmma_bf16(qf[ks], load_frag(p1, p1 + 16), s1);
    }

    // mask: causal && window && same-segment (no 1/sqrt(d) in reference)
#pragma unroll
    for (int r = 0; r < 8; ++r) {
      int row = q0 + r + 8 * half;
      int c0 = kc + ln, c1 = kc + 16 + ln;
      if (!(row >= c0 && (row - c0) < WIN_ && c0 >= seg_start)) s0[r] = -1e30f;
      if (!(row >= c1 && (row - c1) < WIN_ && c1 >= seg_start)) s1[r] = -1e30f;
    }

    // online softmax update (row stats live in lanes of each half)
    float alpha[8];
#pragma unroll
    for (int r = 0; r < 8; ++r) {
      float mx = fmaxf(s0[r], s1[r]);
#pragma unroll
      for (int off = 8; off >= 1; off >>= 1) mx = fmaxf(mx, __shfl_xor(mx, off, 32));
      float mn = fmaxf(mrow[r], mx);
      alpha[r] = __expf(mrow[r] - mn);
      mrow[r] = mn;
      float p0 = __expf(s0[r] - mn);
      float p1 = __expf(s1[r] - mn);
      s0[r] = p0; s1[r] = p1;
      float sum = p0 + p1;
#pragma unroll
      for (int off = 8; off >= 1; off >>= 1) sum += __shfl_xor(sum, off, 32);
      lrow[r] = lrow[r] * alpha[r] + sum;
    }
#pragma unroll
    for (int c = 0; c < 8; ++c)
#pragma unroll
      for (int r = 0; r < 8; ++r) acc[c][r] *= alpha[r];

    // transpose P (C-layout -> A-layout) through LDS as bf16
#pragma unroll
    for (int r = 0; r < 8; ++r) {
      int row = r + 8 * half;
      pls[row * 32 + ln]      = f2bf(s0[r]);
      pls[row * 32 + 16 + ln] = f2bf(s1[r]);
    }
    __syncthreads();
    const unsigned short* pp = &pls[ln * 32 + 8 * half];
    v16bf pf = load_frag(pp, pp + 16);
    __syncthreads();

    // O += P V  (B-fragments contiguous along keys thanks to v^T layout)
#pragma unroll
    for (int c = 0; c < 8; ++c) {
      const unsigned short* vp = vT + ((long)kh * D_ + c * 16 + ln) * T_ + kc + 8 * half;
      acc[c] = wmma_bf16(pf, load_frag(vp, vp + 16), acc[c]);
    }
  }

  // normalize and store O (bf16, [T][H][128])
#pragma unroll
  for (int c = 0; c < 8; ++c)
#pragma unroll
    for (int r = 0; r < 8; ++r) {
      int row = q0 + r + 8 * half;
      ob[((long)row * H_ + h) * D_ + c * 16 + ln] = f2bf(acc[c][r] / lrow[r]);
    }
}

// ---------------- host-side orchestration ----------------
extern "C" void kernel_launch(void* const* d_in, const int* in_sizes, int n_in,
                              void* d_out, int out_size, void* d_ws, size_t ws_size,
                              hipStream_t stream) {
  const float* x    = (const float*)d_in[0];
  const float* cosb = (const float*)d_in[1];
  const float* sinb = (const float*)d_in[2];
  const float* wq0  = (const float*)d_in[3];
  const float* wk0  = (const float*)d_in[4];
  const float* wv0  = (const float*)d_in[5];
  const float* wo0  = (const float*)d_in[6];
  const float* qn0  = (const float*)d_in[7];
  const float* kn0  = (const float*)d_in[8];
  const float* wq1  = (const float*)d_in[9];
  const float* wo1  = (const float*)d_in[10];
  const float* qn1  = (const float*)d_in[11];
  (void)in_sizes; (void)n_in; (void)out_size; (void)ws_size;

  // workspace layout (256B aligned)
  char* base = (char*)d_ws;
  size_t off = 0;
  auto alloc = [&](size_t bytes) -> char* {
    char* p = base + off;
    off = (off + bytes + 255) & ~(size_t)255;
    return p;
  };
  unsigned short* x_bf   = (unsigned short*)alloc((size_t)T_ * HID_ * 2);
  unsigned short* wq0T   = (unsigned short*)alloc((size_t)HID_ * (H_ * D_) * 2);
  unsigned short* wk0T   = (unsigned short*)alloc((size_t)HID_ * (KH_ * D_) * 2);
  unsigned short* wv0T   = (unsigned short*)alloc((size_t)HID_ * (KH_ * D_) * 2);
  unsigned short* wo0T   = (unsigned short*)alloc((size_t)(H_ * D_) * HID_ * 2);
  unsigned short* wq1T   = (unsigned short*)alloc((size_t)HID_ * (H_ * D_) * 2);
  unsigned short* wo1T   = (unsigned short*)alloc((size_t)(H_ * D_) * HID_ * 2);
  float*          projf  = (float*)alloc((size_t)T_ * (H_ * D_) * 4);
  unsigned short* q_bf   = (unsigned short*)alloc((size_t)T_ * H_ * D_ * 2);
  unsigned short* k_bf   = (unsigned short*)alloc((size_t)T_ * KH_ * D_ * 2);
  unsigned short* vT_bf  = (unsigned short*)alloc((size_t)KH_ * D_ * T_ * 2);
  unsigned short* a_bf   = (unsigned short*)alloc((size_t)T_ * H_ * D_ * 2);
  float*          out0f  = (float*)alloc((size_t)T_ * HID_ * 4);
  unsigned short* out0bf = (unsigned short*)alloc((size_t)T_ * HID_ * 2);

  const long nx = (long)T_ * HID_;
  const dim3 blk256(256);

  // 1) cast input activations
  cast_f32_bf16_kernel<<<(nx + 255) / 256, blk256, 0, stream>>>(x, x_bf, nx);

  // 2) transpose+cast all weights to [N][K] bf16
  auto tgrid = [](long k, long n) { return dim3((unsigned)((k * n + 255) / 256)); };
  transpose_cast_kernel<<<tgrid(HID_, H_ * D_),  blk256, 0, stream>>>(wq0, wq0T, HID_, H_ * D_);
  transpose_cast_kernel<<<tgrid(HID_, KH_ * D_), blk256, 0, stream>>>(wk0, wk0T, HID_, KH_ * D_);
  transpose_cast_kernel<<<tgrid(HID_, KH_ * D_), blk256, 0, stream>>>(wv0, wv0T, HID_, KH_ * D_);
  transpose_cast_kernel<<<tgrid(H_ * D_, HID_),  blk256, 0, stream>>>(wo0, wo0T, H_ * D_, HID_);
  transpose_cast_kernel<<<tgrid(HID_, H_ * D_),  blk256, 0, stream>>>(wq1, wq1T, HID_, H_ * D_);
  transpose_cast_kernel<<<tgrid(H_ * D_, HID_),  blk256, 0, stream>>>(wo1, wo1T, H_ * D_, HID_);

  auto ggrid = [](int n, int m) { return dim3((unsigned)(n / 128), (unsigned)(m / 128)); };
  const dim3 rblk(256);

  // ---- layer 0 projections ----
  gemm_bf16_kernel<<<ggrid(H_ * D_, T_), blk256, 0, stream>>>(x_bf, wq0T, projf, T_, H_ * D_, HID_);
  rms_rope_kernel<<<T_ * H_ / 8, rblk, 0, stream>>>(projf, qn0, cosb, sinb, q_bf, H_, 1, 0);

  gemm_bf16_kernel<<<ggrid(KH_ * D_, T_), blk256, 0, stream>>>(x_bf, wk0T, projf, T_, KH_ * D_, HID_);
  rms_rope_kernel<<<T_ * KH_ / 8, rblk, 0, stream>>>(projf, kn0, cosb, sinb, k_bf, KH_, 1, 0);

  gemm_bf16_kernel<<<ggrid(KH_ * D_, T_), blk256, 0, stream>>>(x_bf, wv0T, projf, T_, KH_ * D_, HID_);
  rms_rope_kernel<<<T_ * KH_ / 8, rblk, 0, stream>>>(projf, nullptr, cosb, sinb, vT_bf, KH_, 0, 1);

  // ---- layer 0 attention + output projection ----
  attn_kernel<<<dim3(T_ / 16, H_), dim3(32), 0, stream>>>(q_bf, k_bf, vT_bf, a_bf);
  gemm_bf16_kernel<<<ggrid(HID_, T_), blk256, 0, stream>>>(a_bf, wo0T, out0f, T_, HID_, H_ * D_);

  // ---- layer 1 (k, v reused) ----
  cast_f32_bf16_kernel<<<(nx + 255) / 256, blk256, 0, stream>>>(out0f, out0bf, nx);
  gemm_bf16_kernel<<<ggrid(H_ * D_, T_), blk256, 0, stream>>>(out0bf, wq1T, projf, T_, H_ * D_, HID_);
  rms_rope_kernel<<<T_ * H_ / 8, rblk, 0, stream>>>(projf, qn1, cosb, sinb, q_bf, H_, 1, 0);

  attn_kernel<<<dim3(T_ / 16, H_), dim3(32), 0, stream>>>(q_bf, k_bf, vT_bf, a_bf);
  gemm_bf16_kernel<<<ggrid(HID_, T_), blk256, 0, stream>>>(a_bf, wo1T, (float*)d_out, T_, HID_, H_ * D_);
}
